// ESNReservoir_15779709846183
// MI455X (gfx1250) — matrix operand order repaired and checked by reading, not compiled, not run on hardware
//
#include <hip/hip_runtime.h>

// ---------------------------------------------------------------------------
// ESN reservoir on gfx1250: h_t = tanh(Win x_t + W h_{t-1}); y_t = Wout h_t + b
// B=1, T=2048, I=8, R=4096, O=8
// ---------------------------------------------------------------------------

typedef __attribute__((ext_vector_type(16))) _Float16 v16h;
typedef __attribute__((ext_vector_type(8)))  _Float16 v8h;
typedef __attribute__((ext_vector_type(8)))  float    v8f;

#define RDIM     4096
#define TSTEPS   2048
#define IDIM     8
#define ODIM     8
#define NTILES   256          // RDIM / 16
#define NCHUNK   128          // RDIM / 32 (K per WMMA = 32)
#define NBLOCKS  128          // persistent blocks
#define WPB      2            // waves per block (64 threads)

// ---------------------------------------------------------------------------
// Kernel 0: init workspace state that must be fresh every call
// ---------------------------------------------------------------------------
__global__ void esn_init(_Float16* __restrict__ Hbuf, unsigned int* __restrict__ cnt) {
    int tid = blockIdx.x * blockDim.x + threadIdx.x;
    if (tid < RDIM) Hbuf[tid] = (_Float16)0.0f;   // h_0 = 0 (row 0 of H history)
    if (tid < TSTEPS) cnt[tid] = 0u;               // per-step barrier counters
}

// ---------------------------------------------------------------------------
// Kernel 1: pack W (f32 row-major [R,R]) into fp16 WMMA A-fragment layout.
// Fragment block (tile, c): 512 halfs; lane L owns halfs [L*16, L*16+16):
//   lane <16 : row = tile*16+L,      K = c*32 + {0..15}
//   lane >=16: row = tile*16+(L-16), K = c*32 + {16..31}
// ---------------------------------------------------------------------------
__global__ void esn_pack_w(const float* __restrict__ W, _Float16* __restrict__ Wp) {
    size_t gid = (size_t)blockIdx.x * blockDim.x + threadIdx.x;  // 0 .. 256*128*32-1
    int lane   = (int)(gid & 31);
    size_t blk = gid >> 5;             // tile*128 + c
    int c      = (int)(blk & (NCHUNK - 1));
    int tile   = (int)(blk >> 7);
    int m      = tile * 16 + (lane & 15);
    int kb     = c * 32 + ((lane & 16) ? 16 : 0);
    const float* src = W + (size_t)m * RDIM + kb;
    _Float16* dst = Wp + gid * 16;
#pragma unroll
    for (int j = 0; j < 8; ++j) dst[j]     = (_Float16)src[j];
#pragma unroll
    for (int j = 0; j < 8; ++j) dst[8 + j] = (_Float16)src[8 + j];
}

// ---------------------------------------------------------------------------
// Kernel 2: persistent recurrence. 128 blocks x 64 threads = 256 waves,
// one 16-row tile per wave. Device-wide sync per step via counter cnt[t].
// ---------------------------------------------------------------------------
__device__ __forceinline__ float fast_tanh(float x) {
    float e = __expf(2.0f * x);          // inf-safe: e=inf -> 1, e=0 -> -1
    return 1.0f - 2.0f / (e + 1.0f);
}

__global__ void __launch_bounds__(64)
esn_recur(const float* __restrict__ x, const float* __restrict__ Win,
          const _Float16* __restrict__ Wp, _Float16* __restrict__ Hbuf,
          unsigned int* __restrict__ cnt) {
    const int lane = threadIdx.x & 31;
    const int tile = blockIdx.x * WPB + (threadIdx.x >> 5);     // 0..255
    const _Float16* Abase = Wp + (size_t)tile * NCHUNK * 512 + (size_t)lane * 16;
    const int hoff = (lane & 16) ? 16 : 0;                      // B-frag K half

    for (int t = 0; t < TSTEPS; ++t) {
        const _Float16* hprev = Hbuf + (size_t)t * RDIM;        // h_{t-1}
        v8f acc0 = {}, acc1 = {}, acc2 = {}, acc3 = {};

        for (int c = 0; c < NCHUNK; c += 4) {
            __builtin_prefetch(Abase + (size_t)(c + 8) * 512, 0, 1);
            v16h a0 = *(const v16h*)(Abase + (size_t)(c + 0) * 512);
            v16h a1 = *(const v16h*)(Abase + (size_t)(c + 1) * 512);
            v16h a2 = *(const v16h*)(Abase + (size_t)(c + 2) * 512);
            v16h a3 = *(const v16h*)(Abase + (size_t)(c + 3) * 512);
            v16h b0 = *(const v16h*)(hprev + (c + 0) * 32 + hoff);
            v16h b1 = *(const v16h*)(hprev + (c + 1) * 32 + hoff);
            v16h b2 = *(const v16h*)(hprev + (c + 2) * 32 + hoff);
            v16h b3 = *(const v16h*)(hprev + (c + 3) * 32 + hoff);
            acc0 = __builtin_amdgcn_wmma_f32_16x16x32_f16(false, a0, false, b0,
                                                          (short)0, acc0, false, false);
            acc1 = __builtin_amdgcn_wmma_f32_16x16x32_f16(false, a1, false, b1,
                                                          (short)0, acc1, false, false);
            acc2 = __builtin_amdgcn_wmma_f32_16x16x32_f16(false, a2, false, b2,
                                                          (short)0, acc2, false, false);
            acc3 = __builtin_amdgcn_wmma_f32_16x16x32_f16(false, a3, false, b3,
                                                          (short)0, acc3, false, false);
        }
        v8f acc = (acc0 + acc1) + (acc2 + acc3);

        // All 16 lanes of a half-wave hold identical columns; lane 0 owns rows
        // tile*16+0..7 (acc[0..7]), lane 16 owns rows tile*16+8..15.
        if (lane == 0 || lane == 16) {
            const int rbase = tile * 16 + ((lane == 16) ? 8 : 0);
            const float4 xa = *(const float4*)(x + (size_t)t * IDIM);
            const float4 xb = *(const float4*)(x + (size_t)t * IDIM + 4);
            v8h hout;
#pragma unroll
            for (int j = 0; j < 8; ++j) {
                const float4* wr = (const float4*)(Win + (size_t)(rbase + j) * IDIM);
                float u = wr[0].x * xa.x + wr[0].y * xa.y + wr[0].z * xa.z + wr[0].w * xa.w
                        + wr[1].x * xb.x + wr[1].y * xb.y + wr[1].z * xb.z + wr[1].w * xb.w;
                hout[j] = (_Float16)fast_tanh(acc[j] + u);
            }
            *(v8h*)(Hbuf + (size_t)(t + 1) * RDIM + rbase) = hout;   // h_t
        }

        // ---- device-wide step barrier (per-step monotonic counter) ----
        __syncthreads();
        if (threadIdx.x == 0) {
            __hip_atomic_fetch_add(cnt + t, 1u, __ATOMIC_RELEASE, __HIP_MEMORY_SCOPE_AGENT);
            while (__hip_atomic_load(cnt + t, __ATOMIC_ACQUIRE, __HIP_MEMORY_SCOPE_AGENT)
                   < (unsigned)NBLOCKS) {
                __builtin_amdgcn_s_sleep(1);
            }
            __builtin_amdgcn_fence(__ATOMIC_ACQUIRE, "agent");
        }
        __syncthreads();
    }
}

// ---------------------------------------------------------------------------
// Kernel 3: y_t = Wout h_t + b.  One block per timestep, LDS tree reduction.
// ---------------------------------------------------------------------------
__global__ void esn_out(const _Float16* __restrict__ Hbuf,
                        const float* __restrict__ Wout,
                        const float* __restrict__ bias,
                        float* __restrict__ y) {
    const int t = blockIdx.x;
    const int tid = threadIdx.x;
    const _Float16* h = Hbuf + (size_t)(t + 1) * RDIM;
    float acc[ODIM];
#pragma unroll
    for (int o = 0; o < ODIM; ++o) acc[o] = 0.0f;

    for (int r = tid; r < RDIM; r += 256) {
        float hv = (float)h[r];
#pragma unroll
        for (int o = 0; o < ODIM; ++o)
            acc[o] += Wout[(size_t)o * RDIM + r] * hv;
    }

    __shared__ float s[256 * ODIM];
#pragma unroll
    for (int o = 0; o < ODIM; ++o) s[o * 256 + tid] = acc[o];
    __syncthreads();
    for (int st = 128; st > 0; st >>= 1) {
        if (tid < st) {
#pragma unroll
            for (int o = 0; o < ODIM; ++o)
                s[o * 256 + tid] += s[o * 256 + tid + st];
        }
        __syncthreads();
    }
    if (tid < ODIM) y[(size_t)t * ODIM + tid] = s[tid * 256] + bias[tid];
}

// ---------------------------------------------------------------------------
// Launch: inputs in setup_inputs() order: x, Win, W, Wout_w, Wout_b (all f32)
// ---------------------------------------------------------------------------
extern "C" void kernel_launch(void* const* d_in, const int* in_sizes, int n_in,
                              void* d_out, int out_size, void* d_ws, size_t ws_size,
                              hipStream_t stream) {
    const float* x      = (const float*)d_in[0];   // [1, 2048, 8]
    const float* Win    = (const float*)d_in[1];   // [4096, 8]
    const float* W      = (const float*)d_in[2];   // [4096, 4096]
    const float* Wout_w = (const float*)d_in[3];   // [8, 4096]
    const float* Wout_b = (const float*)d_in[4];   // [8]
    float* y            = (float*)d_out;           // [1, 2048, 8]

    // Workspace layout: Wp fp16 (32 MB) | H history fp16 ((T+1)*R) | counters
    char* ws = (char*)d_ws;
    _Float16*     Wp   = (_Float16*)ws;                                      // 33554432 B
    _Float16*     Hbuf = (_Float16*)(ws + (size_t)RDIM * RDIM * 2);          // 16785408 B
    unsigned int* cnt  = (unsigned int*)(ws + (size_t)RDIM * RDIM * 2
                                            + (size_t)(TSTEPS + 1) * RDIM * 2);

    esn_init<<<16, 256, 0, stream>>>(Hbuf, cnt);

    // pack: 256 tiles * 128 chunks * 32 lanes threads
    esn_pack_w<<<(NTILES * NCHUNK * 32) / 256, 256, 0, stream>>>(W, Wp);

    esn_recur<<<NBLOCKS, WPB * 32, 0, stream>>>(x, Win, Wp, Hbuf, cnt);

    esn_out<<<TSTEPS, 256, 0, stream>>>(Hbuf, Wout_w, Wout_b, y);
}
